// LstmCrfPaSl_10488310136871
// MI455X (gfx1250) — matrix operand order repaired and checked by reading, not compiled, not run on hardware
//
#include <hip/hip_runtime.h>
#include <hip/hip_bf16.h>

// ---------------- problem constants (match reference) ----------------
constexpr int B    = 256;
constexpr int T    = 512;
constexpr int E    = 128;
constexpr int HID  = 256;
constexpr int HD   = 128;      // per-direction hidden
constexpr int G4   = 512;      // 4*HD gate width
constexpr int TAG  = 20;
constexpr int START_IDX = 1;
constexpr int STOP_IDX  = 2;

typedef __attribute__((ext_vector_type(16))) _Float16 v16h;
typedef __attribute__((ext_vector_type(8)))  float    v8f;
typedef __attribute__((ext_vector_type(4)))  int      v4i_;

union AFrag { v16h v; uint4 q[2]; };
union BFrag { v16h v; uint4 q[2]; };

__device__ __forceinline__ float sigmf(float x) { return 1.0f / (1.0f + expf(-x)); }

// CDNA5 async global->LDS path (ASYNCcnt). Guarded: falls back to a
// synchronous vector copy if this toolchain does not declare the builtins.
#if defined(__gfx1250__) && __has_builtin(__builtin_amdgcn_global_load_async_to_lds_b128) && __has_builtin(__builtin_amdgcn_s_wait_asynccnt)
#define USE_ASYNC_LDS 1
#else
#define USE_ASYNC_LDS 0
#endif

__device__ __forceinline__ void stage_b128(const _Float16* src, _Float16* dst_lds) {
#if USE_ASYNC_LDS
    // Builtin signature (from the compiler): (v4i AS1*, v4i AS3*, imm, imm).
    // Flat LDS address: low 32 bits are the LDS offset (aperture in the high
    // bits), so an integer cast yields a valid AS(3) pointer.
    __builtin_amdgcn_global_load_async_to_lds_b128(
        (__attribute__((address_space(1))) v4i_*)(size_t)src,
        (__attribute__((address_space(3))) v4i_*)(unsigned int)(size_t)dst_lds,
        0, 0);
#else
    *(uint4*)dst_lds = *(const uint4*)src;
#endif
}

// ---------------------------------------------------------------------
// Kernel 1: embedding gather (f32 -> f16) + validity mask.
// x layout: [T][B][E] f16 (time-major so an LSTM batch tile is 16
// consecutive rows of 128 contiguous halves). mask layout: [T][B] f32.
// ---------------------------------------------------------------------
__global__ void embed_mask_kernel(const int* __restrict__ chars,
                                  const float* __restrict__ etab,
                                  _Float16* __restrict__ x16,
                                  float* __restrict__ msk) {
    size_t idx = (size_t)blockIdx.x * 256 + threadIdx.x;   // over B*T*E
    int    e   = (int)(idx & (E - 1));
    size_t bt  = idx >> 7;                                  // b*T + t
    int    t   = (int)(bt & (T - 1));
    int    b   = (int)(bt >> 9);
    int    id  = chars[bt];
    float  v   = etab[(size_t)id * E + e];
    x16[((size_t)t * B + b) * E + e] = (_Float16)v;
    if (e == 0) msk[(size_t)t * B + b] = (id > 0) ? 1.0f : 0.0f;
}

// ---------------------------------------------------------------------
// Kernel 2: one bidirectional-LSTM direction per blockIdx.y.
// grid = (B/16, 2), block = 256 threads (8 waves).
// All weights live in LDS for the entire 512-step recurrence (CDNA5:
// 320 KB/WGP). Wave w owns gate columns j0 = 16*w of each of the four
// gate sections (i,f,g,o), so its four WMMA accumulators carry matching
// (batch-row, hidden-unit) elements and the nonlinearity is lane-local.
// Cell state c stays in registers across all timesteps. x tiles are
// double-buffered and streamed with the async global->LDS engine.
// ---------------------------------------------------------------------
#define LSTM_LDS_BYTES 276544
__global__ void lstm_kernel(const _Float16* __restrict__ x,
                            const float* __restrict__ msk,
                            const float* __restrict__ wih_f, const float* __restrict__ whh_f,
                            const float* __restrict__ bih_f, const float* __restrict__ bhh_f,
                            const float* __restrict__ wih_b, const float* __restrict__ whh_b,
                            const float* __restrict__ bih_b, const float* __restrict__ bhh_b,
                            _Float16* __restrict__ h_out /* [T][B][2][HD] f16 */) {
    extern __shared__ char smem[];
    _Float16* Wx     = (_Float16*)(smem);              // [512][128] f16 = 128 KB
    _Float16* Wh     = (_Float16*)(smem + 131072);     // [512][128] f16 = 128 KB
    float*    bias_s = (float*)   (smem + 262144);     // [512] f32
    _Float16* x_s0   = (_Float16*)(smem + 264192);     // [16][128] f16 (buf 0)
    _Float16* x_s1   = (_Float16*)(smem + 268288);     // [16][128] f16 (buf 1)
    _Float16* h_s    = (_Float16*)(smem + 272384);     // [16][128] f16
    float*    m_s    = (float*)   (smem + 276480);     // [16] f32

    const int tid  = threadIdx.x;
    const int lane = tid & 31;
    const int wave = tid >> 5;
    const int b0   = blockIdx.x * 16;
    const int dir  = blockIdx.y;
    const int j0   = wave * 16;

    const float* wih = dir ? wih_b : wih_f;
    const float* whh = dir ? whh_b : whh_f;
    const float* bih = dir ? bih_b : bih_f;
    const float* bhh = dir ? bhh_b : bhh_f;

    // One-time: copy weights (4HD,K) into LDS as f16, same n-major layout.
    for (int idx = tid; idx < 128 * 512; idx += 256) {
        int k = idx & 127;          // 0..127
        int n = idx >> 7;           // 0..511
        Wx[idx] = (_Float16)wih[n * E + k];
        Wh[idx] = (_Float16)whh[n * HD + k];
    }
    for (int n = tid; n < 512; n += 256) bias_s[n] = bih[n] + bhh[n];
    for (int idx = tid; idx < 16 * HD; idx += 256) h_s[idx] = (_Float16)0.0f;

    // Each thread owns one 16-byte slice of the x tile.
    const int xrow_ld = tid >> 4;        // 0..15 (batch row in tile)
    const int xseg_ld = tid & 15;        // 0..15 (8 halves each)
    auto issue_x = [&](int t, _Float16* buf) {
        stage_b128(x + ((size_t)t * B + b0 + xrow_ld) * E + xseg_ld * 8,
                   buf + xrow_ld * E + xseg_ld * 8);
    };

    // Prime the pipeline with the first timestep's tile.
    issue_x(dir ? (T - 1) : 0, x_s0);
    __syncthreads();   // weights/bias/h_s ready

    float cstate[8];
    #pragma unroll
    for (int r = 0; r < 8; ++r) cstate[r] = 0.0f;

    const int arow  = lane & 15;          // A-matrix row (batch row in tile)
    const int khalf = (lane >> 4) * 8;    // K sub-offset for upper half-wave
    const int Mhi   = (lane & 16) ? 8 : 0;
    const int nloc  = lane & 15;

    for (int step = 0; step < T; ++step) {
        int t = dir ? (T - 1 - step) : step;
        const _Float16* xcur = (step & 1) ? x_s1 : x_s0;

        if (tid < 16) m_s[tid] = msk[(size_t)t * B + b0 + tid];
        if (step + 1 < T) {
            // Stream next tile into the other buffer; its last readers all
            // passed the barrier at the end of step-1 before we got here.
            int tn = dir ? (t - 1) : (t + 1);
            issue_x(tn, ((step + 1) & 1) ? x_s1 : x_s0);
#if USE_ASYNC_LDS
            __builtin_amdgcn_s_wait_asynccnt(1);   // this step's tile landed
#endif
        } else {
#if USE_ASYNC_LDS
            __builtin_amdgcn_s_wait_asynccnt(0);
#endif
        }
        if (step + 2 < T) {   // warm L2 two steps ahead
            int tp = dir ? (t - 2) : (t + 2);
            __builtin_prefetch(x + ((size_t)tp * B + b0 + xrow_ld) * E, 0, 0);
        }
        __syncthreads();   // x tile + mask published to all waves

        // A fragments: 4 from x tile (K=0..127), 4 from h state (K=0..127).
        AFrag ax[4], ah[4];
        const _Float16* xrow = xcur + arow * E  + khalf;
        const _Float16* hrow = h_s  + arow * HD + khalf;
        #pragma unroll
        for (int kk = 0; kk < 4; ++kk) {
            ax[kk].q[0] = *(const uint4*)(xrow + kk * 32);
            ax[kk].q[1] = *(const uint4*)(xrow + kk * 32 + 16);
            ah[kk].q[0] = *(const uint4*)(hrow + kk * 32);
            ah[kk].q[1] = *(const uint4*)(hrow + kk * 32 + 16);
        }

        v8f acc[4];   // gate order i,f,g,o at columns sec*HD + j0 + n
        #pragma unroll
        for (int sec = 0; sec < 4; ++sec) {
            float bv = bias_s[sec * HD + j0 + nloc];
            v8f a;
            #pragma unroll
            for (int r = 0; r < 8; ++r) a[r] = bv;
            #pragma unroll
            for (int kk = 0; kk < 4; ++kk) {
                BFrag bw;
                const _Float16* p = Wx + (size_t)(sec * HD + j0 + nloc) * 128 + kk * 32 + khalf;
                bw.q[0] = *(const uint4*)(p);
                bw.q[1] = *(const uint4*)(p + 16);
                a = __builtin_amdgcn_wmma_f32_16x16x32_f16(false, ax[kk].v, false, bw.v,
                                                           (short)0, a, false, false);
            }
            #pragma unroll
            for (int kk = 0; kk < 4; ++kk) {
                BFrag bw;
                const _Float16* p = Wh + (size_t)(sec * HD + j0 + nloc) * 128 + kk * 32 + khalf;
                bw.q[0] = *(const uint4*)(p);
                bw.q[1] = *(const uint4*)(p + 16);
                a = __builtin_amdgcn_wmma_f32_16x16x32_f16(false, ah[kk].v, false, bw.v,
                                                           (short)0, a, false, false);
            }
            acc[sec] = a;
        }

        // Lane-local LSTM cell update (PyTorch gate order i,f,g,o).
        float hn[8];
        #pragma unroll
        for (int r = 0; r < 8; ++r) {
            float m  = m_s[r + Mhi];
            float ig = sigmf(acc[0][r]);
            float fg = sigmf(acc[1][r]);
            float gg = tanhf(acc[2][r]);
            float og = sigmf(acc[3][r]);
            float cn = fg * cstate[r] + ig * gg;
            float hv = og * tanhf(cn);
            cstate[r] = cn * m;       // masked state (packed-seq semantics)
            hn[r]     = hv * m;
        }
        __syncthreads();   // everyone done reading h_s / x tiles
        #pragma unroll
        for (int r = 0; r < 8; ++r) {
            int M = r + Mhi;
            _Float16 hv = (_Float16)hn[r];
            h_s[M * HD + j0 + nloc] = hv;
            h_out[(((size_t)t * B + b0 + M) * 2 + dir) * HD + j0 + nloc] = hv;
        }
    }
}

// ---------------------------------------------------------------------
// Kernel 3: emissions. rows = T*B, K = HID = 256, N = TAG padded to 32.
// grid = T*B/128 blocks, 256 threads (8 waves x 16 rows each).
// ---------------------------------------------------------------------
__global__ void emission_kernel(const _Float16* __restrict__ h_out,
                                const float* __restrict__ W_lin,
                                const float* __restrict__ b_lin,
                                float* __restrict__ score /* [T*B][TAG] */) {
    __shared__ _Float16 WT[256 * 32];  // [N=32][K=256] f16, rows n>=20 zero
    __shared__ float    bl[32];

    const int tid = threadIdx.x, lane = tid & 31, wave = tid >> 5;
    for (int idx = tid; idx < 256 * 32; idx += 256) {
        int k = idx & 255, n = idx >> 8;
        WT[idx] = (n < TAG) ? (_Float16)W_lin[n * HID + k] : (_Float16)0.0f;
    }
    if (tid < 32) bl[tid] = (tid < TAG) ? b_lin[tid] : 0.0f;
    __syncthreads();

    const size_t row0  = ((size_t)blockIdx.x * 8 + wave) * 16;
    const int    arow  = lane & 15;
    const int    khalf = (lane >> 4) * 8;
    const int    Mhi   = (lane & 16) ? 8 : 0;
    const int    nloc  = lane & 15;
    const _Float16* rp = h_out + (row0 + arow) * HID;

    v8f acc0, acc1;
    {
        float b0v = bl[nloc], b1v = bl[16 + nloc];
        #pragma unroll
        for (int r = 0; r < 8; ++r) { acc0[r] = b0v; acc1[r] = b1v; }
    }
    #pragma unroll
    for (int kk = 0; kk < 8; ++kk) {
        AFrag a;
        const _Float16* p = rp + kk * 32 + khalf;
        a.q[0] = *(const uint4*)(p);
        a.q[1] = *(const uint4*)(p + 16);
        BFrag w0, w1;
        const _Float16* q = WT + (size_t)nloc * 256 + kk * 32 + khalf;
        w0.q[0] = *(const uint4*)(q);             w0.q[1] = *(const uint4*)(q + 16);
        w1.q[0] = *(const uint4*)(q + 16 * 256); w1.q[1] = *(const uint4*)(q + 16 * 256 + 16);
        acc0 = __builtin_amdgcn_wmma_f32_16x16x32_f16(false, a.v, false, w0.v,
                                                      (short)0, acc0, false, false);
        acc1 = __builtin_amdgcn_wmma_f32_16x16x32_f16(false, a.v, false, w1.v,
                                                      (short)0, acc1, false, false);
    }
    #pragma unroll
    for (int r = 0; r < 8; ++r) {
        size_t row = row0 + r + Mhi;
        score[row * TAG + nloc] = acc0[r];
    }
    if (nloc < TAG - 16) {
        #pragma unroll
        for (int r = 0; r < 8; ++r) {
            size_t row = row0 + r + Mhi;
            score[row * TAG + 16 + nloc] = acc1[r];
        }
    }
}

// ---------------------------------------------------------------------
// Kernel 4: CRF forward (log-partition) + partial-annotation gold score.
// One wave per batch element; lane l < 20 owns current-tag l.
// ---------------------------------------------------------------------
__global__ void crf_kernel(const float* __restrict__ score /* [T*B][TAG] */,
                           const float* __restrict__ msk   /* [T][B]    */,
                           const float* __restrict__ tagm  /* [B][T][TAG] */,
                           const float* __restrict__ trans /* [TAG][TAG] */,
                           float* __restrict__ partial) {
    __shared__ float tr[TAG * TAG];
    __shared__ float a_sh[TAG], g_sh[TAG], mp_sh[TAG], e_sh[TAG], mc_sh[TAG];

    const int b = blockIdx.x;
    const int l = threadIdx.x;
    for (int idx = l; idx < TAG * TAG; idx += 32) tr[idx] = trans[idx];
    __syncthreads();

    float alpha = 0.0f, gold = 0.0f, mask_pre = 0.0f;
    if (l < TAG) {
        float e0 = score[(size_t)b * TAG + l];
        alpha    = e0 + tr[l * TAG + START_IDX];
        mask_pre = tagm[((size_t)b * T) * TAG + l];
        gold     = alpha * mask_pre;
    }
    for (int t = 1; t < T; ++t) {
        __syncthreads();
        if (l < TAG) {
            a_sh[l]  = alpha;
            g_sh[l]  = gold;
            mp_sh[l] = mask_pre;
            e_sh[l]  = score[((size_t)t * B + b) * TAG + l];
            mc_sh[l] = tagm[((size_t)b * T + t) * TAG + l];
        }
        __syncthreads();
        float mt = msk[(size_t)t * B + b];
        if (l < TAG) {
            // forward: alpha'[l] = emit[l] + lse_p(alpha[p] + trans[l][p])
            float mx = -3.0e38f;
            for (int p = 0; p < TAG; ++p) mx = fmaxf(mx, a_sh[p] + tr[l * TAG + p]);
            float s = 0.0f;
            for (int p = 0; p < TAG; ++p) s += expf(a_sh[p] + tr[l * TAG + p] - mx);
            float anew = e_sh[l] + mx + logf(s);
            if (mt > 0.0f) alpha = anew;
            // gold: masked lse over allowed (prev,cur) pairs
            float mcl = mc_sh[l];
            float gmx = -3.0e38f;
            int   any = 0;
            if (mcl > 0.0f) {
                for (int p = 0; p < TAG; ++p)
                    if (mp_sh[p] > 0.0f) { gmx = fmaxf(gmx, g_sh[p] + tr[l * TAG + p]); any = 1; }
            }
            float st = 0.0f;
            if (any) {
                float gs = 0.0f;
                for (int p = 0; p < TAG; ++p)
                    if (mp_sh[p] > 0.0f) gs += expf(g_sh[p] + tr[l * TAG + p] - gmx);
                st = e_sh[l] + gmx + logf(gs);
            }
            if (mt > 0.0f) gold = st;
            mask_pre = mcl;   // unconditional carry, as in reference
        }
    }
    __syncthreads();
    if (l < TAG) {
        a_sh[l]  = alpha + tr[STOP_IDX * TAG + l];
        float ms = (gold != 0.0f) ? 1.0f : 0.0f;
        g_sh[l]  = gold + tr[STOP_IDX * TAG + l] * ms;
        mp_sh[l] = ms;
    }
    __syncthreads();
    if (l == 0) {
        float mx = -3.0e38f;
        for (int i = 0; i < TAG; ++i) mx = fmaxf(mx, a_sh[i]);
        float s = 0.0f;
        for (int i = 0; i < TAG; ++i) s += expf(a_sh[i] - mx);
        float fs = mx + logf(s);
        float gmx = -3.0e38f; int any = 0;
        for (int i = 0; i < TAG; ++i)
            if (mp_sh[i] > 0.0f) { gmx = fmaxf(gmx, g_sh[i]); any = 1; }
        float gs = 0.0f;
        if (any) {
            float acc = 0.0f;
            for (int i = 0; i < TAG; ++i)
                if (mp_sh[i] > 0.0f) acc += expf(g_sh[i] - gmx);
            gs = gmx + logf(acc);
        }
        partial[b] = fs - gs;
    }
}

// ---------------------------------------------------------------------
// Kernel 5: deterministic tree reduction of the B per-sequence losses.
// ---------------------------------------------------------------------
__global__ void reduce_kernel(const float* __restrict__ partial, float* __restrict__ out) {
    __shared__ float s[256];
    int tid = threadIdx.x;
    s[tid] = partial[tid];
    __syncthreads();
    for (int off = 128; off > 0; off >>= 1) {
        if (tid < off) s[tid] += s[tid + off];
        __syncthreads();
    }
    if (tid == 0) out[0] = s[0];
}

// ---------------------------------------------------------------------
extern "C" void kernel_launch(void* const* d_in, const int* in_sizes, int n_in,
                              void* d_out, int out_size, void* d_ws, size_t ws_size,
                              hipStream_t stream) {
    (void)in_sizes; (void)n_in; (void)out_size; (void)ws_size;
    const int*   chars = (const int*)d_in[0];
    const float* tagm  = (const float*)d_in[1];
    const float* etab  = (const float*)d_in[2];
    const float* wih_f = (const float*)d_in[3];
    const float* whh_f = (const float*)d_in[4];
    const float* bih_f = (const float*)d_in[5];
    const float* bhh_f = (const float*)d_in[6];
    const float* wih_b = (const float*)d_in[7];
    const float* whh_b = (const float*)d_in[8];
    const float* bih_b = (const float*)d_in[9];
    const float* bhh_b = (const float*)d_in[10];
    const float* W_lin = (const float*)d_in[11];
    const float* b_lin = (const float*)d_in[12];
    const float* trans = (const float*)d_in[13];
    float* out = (float*)d_out;

    char* ws = (char*)d_ws;
    size_t o = 0;
    _Float16* x16   = (_Float16*)(ws + o); o += (size_t)T * B * E * 2;     //  32 MB
    _Float16* hout  = (_Float16*)(ws + o); o += (size_t)T * B * HID * 2;   //  64 MB
    float*    score = (float*)   (ws + o); o += (size_t)T * B * TAG * 4;   //  10 MB
    float*    msk   = (float*)   (ws + o); o += (size_t)T * B * 4;         // 0.5 MB
    float*    part  = (float*)   (ws + o);                                  //  1 KB

    embed_mask_kernel<<<(B * T * E) / 256, 256, 0, stream>>>(chars, etab, x16, msk);

    dim3 lstm_grid(B / 16, 2);
    lstm_kernel<<<lstm_grid, 256, LSTM_LDS_BYTES, stream>>>(
        x16, msk, wih_f, whh_f, bih_f, bhh_f, wih_b, whh_b, bih_b, bhh_b, hout);

    emission_kernel<<<(T * B) / 128, 256, 0, stream>>>(hout, W_lin, b_lin, score);

    crf_kernel<<<B, 32, 0, stream>>>(score, msk, tagm, trans, part);

    reduce_kernel<<<1, 256, 0, stream>>>(part, out);
}
